// GAPP_76948634075857
// MI455X (gfx1250) — matrix operand order repaired and checked by reading, not compiled, not run on hardware
//
#include <hip/hip_runtime.h>
#include <hip/hip_bf16.h>

typedef _Float16 v16h __attribute__((ext_vector_type(16)));
typedef _Float16 v8h  __attribute__((ext_vector_type(8)));
typedef _Float16 v4h  __attribute__((ext_vector_type(4)));
typedef float    v8f  __attribute__((ext_vector_type(8)));
typedef float    v4f  __attribute__((ext_vector_type(4)));

#define N_NODES 100000
#define N_EDGES 1600000
#define IN_CH   512
#define HID_CH  256
#define OUT_CH  32
#define K_HOPS  5
#define ALPHA   0.5f

// ---------------------------------------------------------------------------
// Pre-swizzle W1 (512x256) and W2 (256x32) into WMMA-B fragment-linear f16:
// for tile (kt,nt), lane l reads 16 contiguous f16 (32B):
//   n = nt*16 + (l&15),  k = kt*32 + (l>>4)*8 + (j&7) + (j>>3)*16   (j = 0..15)
// ---------------------------------------------------------------------------
__global__ void GAPP_prep_weights(const float* __restrict__ W1,
                                  const float* __restrict__ W2,
                                  _Float16* __restrict__ W1s,
                                  _Float16* __restrict__ W2s) {
    int o = blockIdx.x * blockDim.x + threadIdx.x;
    const int NW1 = IN_CH * HID_CH;          // 131072
    if (o < NW1) {
        int j  = o & 15;
        int ln = (o >> 4) & 31;
        int nt = (o >> 9) & 15;              // 16 N-tiles
        int kt = o >> 13;                    // 16 K-tiles
        int n  = nt * 16 + (ln & 15);
        int k  = kt * 32 + (ln >> 4) * 8 + (j & 7) + (j >> 3) * 16;
        W1s[o] = (_Float16)W1[k * HID_CH + n];
    } else {
        int o2 = o - NW1;
        if (o2 < HID_CH * OUT_CH) {          // 8192
            int j  = o2 & 15;
            int ln = (o2 >> 4) & 31;
            int nt = (o2 >> 9) & 1;          // 2 N-tiles
            int kt = o2 >> 10;               // 8 K-tiles
            int n  = nt * 16 + (ln & 15);
            int k  = kt * 32 + (ln >> 4) * 8 + (j & 7) + (j >> 3) * 16;
            W2s[o2] = (_Float16)W2[k * OUT_CH + n];
        }
    }
}

__device__ __forceinline__ v16h GAPP_frag_a(const _Float16* rowp, int k0) {
    // rowp = base of this lane's A row, already offset by (half*8)
    v8h lo = *(const v8h*)(rowp + k0);        // K: k0 + half*8 + 0..7
    v8h hi = *(const v8h*)(rowp + k0 + 16);   // K: k0 + 16 + half*8 + 0..7
    return __builtin_shufflevector(lo, hi, 0,1,2,3,4,5,6,7,8,9,10,11,12,13,14,15);
}

// ---------------------------------------------------------------------------
// Fused MLP: h0 = relu(x@W1+b1)@W2 + b2 for a 16-row node block.
// 256 threads = 8 waves. GEMM1: wave w owns cols [w*32, w*32+32). GEMM2:
// wave w: ntile = w&1, K-segment = (w>>1)*64; LDS reduction of partials.
// ---------------------------------------------------------------------------
__global__ __launch_bounds__(256)
void GAPP_mlp_kernel(const float* __restrict__ x,
                     const _Float16* __restrict__ W1s,
                     const float* __restrict__ b1,
                     const _Float16* __restrict__ W2s,
                     const float* __restrict__ b2,
                     float* __restrict__ h0out) {
    __shared__ alignas(16) unsigned char smem[16640 + 8192];
    _Float16* xs   = (_Float16*)smem;           // [16][520] f16 (x tile)
    _Float16* h1s  = (_Float16*)smem;           // reuse: [16][264] f16 (relu(h1))
    float*    part = (float*)(smem + 16640);    // [8][16][16] partials

    const int t    = threadIdx.x;
    const int r0   = blockIdx.x * 16;
    const int wave = t >> 5;
    const int lane = t & 31;
    const int kh   = lane >> 4;     // lane half
    const int lm   = lane & 15;

    // ---- stage x[r0..r0+15][0..511] -> LDS f16, padded stride 520 ----
    for (int i = 0; i < 8; ++i) {
        int idx4 = t + i * 256;                 // 2048 float4 chunks
        int row  = idx4 >> 7;
        int c4   = (idx4 & 127) << 2;
        v4f xv = *(const v4f*)(x + (size_t)(r0 + row) * IN_CH + c4);
        v4h hv;
        hv.x = (_Float16)xv.x; hv.y = (_Float16)xv.y;
        hv.z = (_Float16)xv.z; hv.w = (_Float16)xv.w;
        *(v4h*)(xs + row * 520 + c4) = hv;
    }
    __syncthreads();

    // ---- GEMM1: 16x256 = sum_K 16x32 @ 32x256 via WMMA f16 ----
    v8f acc0 = {}; v8f acc1 = {};
    const _Float16* arow = xs + lm * 520 + kh * 8;
    for (int kt = 0; kt < 16; ++kt) {
        v16h a  = GAPP_frag_a(arow, kt * 32);
        v16h b0 = *(const v16h*)(W1s + (((kt * 16) + wave * 2 + 0) * 32 + lane) * 16);
        v16h bb = *(const v16h*)(W1s + (((kt * 16) + wave * 2 + 1) * 32 + lane) * 16);
        acc0 = __builtin_amdgcn_wmma_f32_16x16x32_f16(false, a, false, b0, (short)0, acc0, false, false);
        acc1 = __builtin_amdgcn_wmma_f32_16x16x32_f16(false, a, false, bb, (short)0, acc1, false, false);
    }
    __syncthreads();   // xs dead; safe to overwrite with h1s

    // ---- bias + relu -> LDS f16 (stride 264) ----
    {
        int ng0 = wave * 32 + lm;
        int ng1 = ng0 + 16;
        float bb0 = b1[ng0], bb1 = b1[ng1];
        for (int v = 0; v < 8; ++v) {
            int m = v + kh * 8;                 // C/D layout: lane half selects M block
            float f0 = acc0[v] + bb0; f0 = f0 > 0.f ? f0 : 0.f;
            float f1 = acc1[v] + bb1; f1 = f1 > 0.f ? f1 : 0.f;
            h1s[m * 264 + ng0] = (_Float16)f0;
            h1s[m * 264 + ng1] = (_Float16)f1;
        }
    }
    __syncthreads();

    // ---- GEMM2: 16x32 = 16x256 @ 256x32, K split across waves ----
    v8f acc2 = {};
    const int nt2 = wave & 1;
    const int ks  = wave >> 1;                  // 0..3 -> K segment of 64
    const _Float16* a2row = h1s + lm * 264 + kh * 8;
    for (int kk = 0; kk < 2; ++kk) {
        int kt = ks * 2 + kk;                   // 0..7
        v16h a = GAPP_frag_a(a2row, kt * 32);
        v16h b = *(const v16h*)(W2s + ((kt * 2 + nt2) * 32 + lane) * 16);
        acc2 = __builtin_amdgcn_wmma_f32_16x16x32_f16(false, a, false, b, (short)0, acc2, false, false);
    }
    for (int v = 0; v < 8; ++v) {
        int m = v + kh * 8;
        part[wave * 256 + m * 16 + lm] = acc2[v];
    }
    __syncthreads();

    // ---- reduce 4 K-partials per output tile, add b2, write h0 ----
    for (int rep = 0; rep < 2; ++rep) {
        int o   = t + rep * 256;                // 512 outputs
        int nt  = o >> 8;
        int rem = o & 255;
        int m   = rem >> 4;
        int n   = rem & 15;
        float s = b2[nt * 16 + n];
        for (int w = nt; w < 8; w += 2)
            s += part[w * 256 + m * 16 + n];
        h0out[(size_t)(r0 + m) * OUT_CH + nt * 16 + n] = s;
    }
}

// ---------------------------------------------------------------------------
// Degree / propagation kernels
// ---------------------------------------------------------------------------
__global__ void GAPP_deg_init(float* deg) {
    int i = blockIdx.x * 256 + threadIdx.x;
    if (i < N_NODES) deg[i] = 1.0f;             // self loop
}

__global__ void GAPP_deg_edges(const int* __restrict__ dst, float* deg) {
    int e = blockIdx.x * 256 + threadIdx.x;
    if (e < N_EDGES) atomicAdd(&deg[dst[e]], 1.0f);
}

__global__ void GAPP_deg_finish(float* deg) {
    int i = blockIdx.x * 256 + threadIdx.x;
    if (i < N_NODES) deg[i] = rsqrtf(deg[i]);   // deg >= 1 always
}

// hout[i,c] = alpha*h0 + (1-alpha) * dinv[i]^2 * hin[i,c]   (self-loop term)
__global__ void GAPP_self_kernel(const float* __restrict__ h0,
                                 const float* __restrict__ hin,
                                 const float* __restrict__ dinv,
                                 float* __restrict__ hout) {
    int i = blockIdx.x * 256 + threadIdx.x;     // exact: N_NODES*32 % 256 == 0
    int node = i >> 5;
    float di = dinv[node];
    hout[i] = ALPHA * h0[i] + (1.0f - ALPHA) * di * di * hin[i];
}

// scatter: hout[dst] += (1-alpha)*dinv[s]*dinv[d]*hin[src];  one wave per edge
__global__ void GAPP_edge_kernel(const int* __restrict__ edges,
                                 const float* __restrict__ dinv,
                                 const float* __restrict__ hin,
                                 float* __restrict__ hout) {
    int tid = blockIdx.x * 256 + threadIdx.x;   // exact: N_EDGES*32 % 256 == 0
    int e = tid >> 5;
    int c = tid & 31;
    int s = edges[e];
    int d = edges[N_EDGES + e];
    float w = (1.0f - ALPHA) * dinv[s] * dinv[d];
    atomicAdd(&hout[d * OUT_CH + c], w * hin[s * OUT_CH + c]);
}

// ---------------------------------------------------------------------------
extern "C" void kernel_launch(void* const* d_in, const int* in_sizes, int n_in,
                              void* d_out, int out_size, void* d_ws, size_t ws_size,
                              hipStream_t stream) {
    const float* x  = (const float*)d_in[0];
    const float* W1 = (const float*)d_in[1];
    const float* b1 = (const float*)d_in[2];
    const float* W2 = (const float*)d_in[3];
    const float* b2 = (const float*)d_in[4];
    const int* edges = (const int*)d_in[5];
    float* out = (float*)d_out;

    char* ws = (char*)d_ws;
    size_t off = 0;
    auto alloc = [&](size_t bytes) -> void* {
        void* p = ws + off;
        off += (bytes + 255) & ~(size_t)255;
        return p;
    };
    float*    h0   = (float*)   alloc((size_t)N_NODES * OUT_CH * sizeof(float));
    float*    hA   = (float*)   alloc((size_t)N_NODES * OUT_CH * sizeof(float));
    float*    hB   = (float*)   alloc((size_t)N_NODES * OUT_CH * sizeof(float));
    float*    dinv = (float*)   alloc((size_t)N_NODES * sizeof(float));
    _Float16* W1s  = (_Float16*)alloc((size_t)IN_CH * HID_CH * sizeof(_Float16));
    _Float16* W2s  = (_Float16*)alloc((size_t)HID_CH * OUT_CH * sizeof(_Float16));

    // weights -> fragment-linear f16
    GAPP_prep_weights<<<(IN_CH*HID_CH + HID_CH*OUT_CH + 255) / 256, 256, 0, stream>>>(
        W1, W2, W1s, W2s);

    // fused WMMA MLP -> h0
    GAPP_mlp_kernel<<<N_NODES / 16, 256, 0, stream>>>(x, W1s, b1, W2s, b2, h0);

    // symmetric GCN normalization degrees
    GAPP_deg_init  <<<(N_NODES + 255) / 256, 256, 0, stream>>>(dinv);
    GAPP_deg_edges <<<(N_EDGES + 255) / 256, 256, 0, stream>>>(edges + N_EDGES, dinv);
    GAPP_deg_finish<<<(N_NODES + 255) / 256, 256, 0, stream>>>(dinv);

    // APPNP hops (last hop writes d_out directly)
    const float* hin = h0;
    float* pp[2] = { hA, hB };
    for (int hop = 0; hop < K_HOPS; ++hop) {
        float* hout = (hop == K_HOPS - 1) ? out : pp[hop & 1];
        GAPP_self_kernel<<<(N_NODES * OUT_CH) / 256, 256, 0, stream>>>(h0, hin, dinv, hout);
        GAPP_edge_kernel<<<(N_EDGES * OUT_CH) / 256, 256, 0, stream>>>(edges, dinv, hin, hout);
        hin = hout;
    }
}
